// UNetDeltaNet_84679575208522
// MI455X (gfx1250) — matrix operand order repaired and checked by reading, not compiled, not run on hardware
//
#include <hip/hip_runtime.h>
#include <cstdint>
#include <cstddef>

// ---------------------------------------------------------------------------
// UNet DeltaNet block for gfx1250 (MI455X): wave32, v_wmma_f32_16x16x32_f16,
// async global->LDS staging (ASYNCcnt), 128-bit LDS fragment feeds.
// ---------------------------------------------------------------------------

typedef __attribute__((ext_vector_type(16))) _Float16 v16h;
typedef __attribute__((ext_vector_type(8)))  float    v8f;

#define BQ   8
#define CDIM 768
#define LQ   4096      // 64*64 tokens
#define NH   12
#define HD   64
#define NCHK 64        // LQ / 64

__device__ __forceinline__ v8f wmma16(v16h a, v16h b, v8f c) {
  return __builtin_amdgcn_wmma_f32_16x16x32_f16(false, a, false, b, (short)0, c,
                                                false, false);
}

// A fragment (16x32): lanes 0-15/16-31 hold M=0..15; element e maps to
// K = (e&7) + ((e>>3)<<4) + (lane>=16 ? 8 : 0). With cs==1 the 16 halves are
// two contiguous 8-half runs -> two ds_load_b128 per lane.
__device__ __forceinline__ v16h load_a_frag(const _Float16* s, int m0, int k0,
                                            int rs, int cs) {
  const int l  = threadIdx.x & 31;
  const int m  = m0 + (l & 15);
  const int hi = (l >> 4) << 3;
  v16h a;
#pragma unroll
  for (int e = 0; e < 16; ++e) {
    int k = k0 + (e & 7) + ((e >> 3) << 4) + hi;
    a[e] = s[m * rs + k * cs];
  }
  return a;
}

// B fragment (32x16): lane%16 = N column, element e maps to
// K = (e&7) + ((e>>3)<<4) + (lane>=16 ? 8 : 0) (same as A).
__device__ __forceinline__ v16h load_b_frag(const _Float16* s, int k0, int n0,
                                            int rs, int cs) {
  const int l = threadIdx.x & 31;
  const int n  = n0 + (l & 15);
  const int hi = (l >> 4) << 3;
  v16h b;
#pragma unroll
  for (int e = 0; e < 16; ++e) {
    int k = k0 + (e & 7) + ((e >> 3) << 4) + hi;
    b[e] = s[k * rs + n * cs];
  }
  return b;
}

__device__ __forceinline__ float dsigmoid(float v) { return 1.f / (1.f + __expf(-v)); }

// gfx1250 async global->LDS copy (tracked by ASYNCcnt). VDST = LDS byte
// address (low 32 bits of generic LDS pointer), VADDR = 64-bit global address.
__device__ __forceinline__ void async_copy_b128(void* lds_ptr, const void* gptr) {
  unsigned lds = (unsigned)(uintptr_t)lds_ptr;
  asm volatile("global_load_async_to_lds_b128 %0, %1, off"
               :: "v"(lds), "v"(gptr) : "memory");
}
__device__ __forceinline__ void async_wait0() {
  asm volatile("s_wait_asynccnt 0x0" ::: "memory");
}

// ---------------------------------------------------------------------------
// fp32 -> f16 weight conversion
// ---------------------------------------------------------------------------
__global__ __launch_bounds__(256) void cvt_kernel(const float* __restrict__ src,
                                                  _Float16* __restrict__ dst, int n) {
  int i = blockIdx.x * 256 + threadIdx.x;
  if (i < n) dst[i] = (_Float16)src[i];
}

// ---------------------------------------------------------------------------
// Q/K/V projection: per block: 64 tokens x one head (64 channels).
// q,k: silu -> l2norm ; v: silu. Outputs f16 [B,H,L,D].
// ---------------------------------------------------------------------------
__global__ __launch_bounds__(128) void proj_kernel(
    const float* __restrict__ x,
    const _Float16* __restrict__ Wqh, const _Float16* __restrict__ Wkh,
    const _Float16* __restrict__ Wvh,
    _Float16* __restrict__ Qh, _Float16* __restrict__ Kh, _Float16* __restrict__ Vh) {
  __shared__ __align__(16) _Float16 As[64 * 32];   // [m][k] row-major
  __shared__ __align__(16) _Float16 Bs[32 * 64];   // [k][n]
  __shared__ float    Cs[64 * 64];                 // result staging

  const int b   = blockIdx.x >> 6;
  const int l0  = (blockIdx.x & 63) << 6;
  const int h   = blockIdx.y;
  const int z   = blockIdx.z;              // 0=Q 1=K 2=V
  const int n0g = h * HD;
  const int tid = threadIdx.x, wave = tid >> 5, lane = tid & 31;
  const int n_l = lane & 15, mbase = wave * 16 + (lane >> 4) * 8;

  const _Float16* W = (z == 0) ? Wqh : (z == 1) ? Wkh : Wvh;
  _Float16* dst     = (z == 0) ? Qh  : (z == 1) ? Kh  : Vh;
  const float* xb   = x + (size_t)b * CDIM * LQ;

  v8f zero = {};
  v8f acc[4] = {zero, zero, zero, zero};

  const int am  = tid & 63;          // token row this thread stages
  const int akb = (tid >> 6) * 16;   // 16 k's per thread

  for (int k0 = 0; k0 < CDIM; k0 += 32) {
    // A tile: coalesced strided global reads, vector LDS writes ([m][k]).
    {
      __align__(16) _Float16 hv[16];
#pragma unroll
      for (int j = 0; j < 16; ++j)
        hv[j] = (_Float16)xb[(size_t)(k0 + akb + j) * LQ + l0 + am];
      *(uint4*)&As[am * 32 + akb]     = ((const uint4*)hv)[0];
      *(uint4*)&As[am * 32 + akb + 8] = ((const uint4*)hv)[1];
    }
    // B tile: straight 128-bit copies.
    {
      const uint4* wsrc = (const uint4*)(W + (size_t)k0 * CDIM + n0g);
      uint4* bdst = (uint4*)Bs;
      for (int i = tid; i < 256; i += 128) {
        int kk = i >> 3, j = i & 7;
        bdst[i] = wsrc[(size_t)kk * (CDIM / 8) + j];
      }
    }
    __syncthreads();
    v16h a = load_a_frag(As, wave * 16, 0, /*rs=*/32, /*cs=*/1);
#pragma unroll
    for (int nt = 0; nt < 4; ++nt)
      acc[nt] = wmma16(a, load_b_frag(Bs, 0, nt * 16, 64, 1), acc[nt]);
    __syncthreads();
  }

  // silu into LDS
#pragma unroll
  for (int nt = 0; nt < 4; ++nt)
#pragma unroll
    for (int r = 0; r < 8; ++r) {
      float v = acc[nt][r];
      Cs[(mbase + r) * 64 + nt * 16 + n_l] = v * dsigmoid(v);
    }
  __syncthreads();

  if (z < 2) {  // l2norm per token row
    if (tid < 64) {
      float ss = 0.f;
      for (int d = 0; d < 64; ++d) { float v = Cs[tid * 64 + d]; ss += v * v; }
      float sc = rsqrtf(ss + 1e-6f);
      for (int d = 0; d < 64; ++d) Cs[tid * 64 + d] *= sc;
    }
    __syncthreads();
  }

  // vectorized f16 store: 8 halves per 128-bit store
  _Float16* dp = dst + ((size_t)(b * NH + h) * LQ + l0) * HD;
  for (int i = tid; i < 512; i += 128) {
    int m = i >> 3, jb = (i & 7) * 8;
    __align__(16) _Float16 tmp[8];
#pragma unroll
    for (int j = 0; j < 8; ++j) tmp[j] = (_Float16)Cs[m * 64 + jb + j];
    *(uint4*)&dp[(size_t)m * HD + jb] = *(const uint4*)tmp;
  }
}

// ---------------------------------------------------------------------------
// beta = sigmoid(t @ Wb)  -> f32 [B,H,L]
// ---------------------------------------------------------------------------
__global__ __launch_bounds__(256) void beta_kernel(const float* __restrict__ x,
                                                   const float* __restrict__ Wb,
                                                   float* __restrict__ Beta) {
  const int l = blockIdx.x * 256 + threadIdx.x;
  const int h = blockIdx.y, b = blockIdx.z;
  const float* xb = x + (size_t)b * CDIM * LQ;
  float s = 0.f;
  for (int c = 0; c < CDIM; ++c) s += xb[(size_t)c * LQ + l] * Wb[c * NH + h];
  Beta[((size_t)b * NH + h) * LQ + l] = dsigmoid(s);
}

// ---------------------------------------------------------------------------
// Chunkwise delta rule. One workgroup per (b,h). S held in LDS (f32 + f16).
// Chunk tiles staged with async global->LDS copies (ASYNCcnt).
// ---------------------------------------------------------------------------
__global__ __launch_bounds__(128) void delta_kernel(
    const _Float16* __restrict__ Qh, const _Float16* __restrict__ Kh,
    const _Float16* __restrict__ Vh, const float* __restrict__ Beta,
    _Float16* __restrict__ Oh) {
  __shared__ float    Sf[64 * 64];               // state accumulator (f32)
  __shared__ _Float16 Sh[64 * 64];               // state mirror (f16)
  __shared__ __align__(16) _Float16 Qis[64 * 64];   // [tok][d]
  __shared__ __align__(16) _Float16 Kis[64 * 64];   // [tok][d]
  __shared__ __align__(16) _Float16 KisT[64 * 64];  // [d][tok]
  __shared__ __align__(16) _Float16 Vis[64 * 64];   // [tok][d]
  __shared__ float    Tm[64 * 64];               // tri system / O staging
  __shared__ float    Rm[64 * 64];               // rhs -> U (in place)
  __shared__ __align__(16) _Float16 Uh[64 * 64];
  __shared__ __align__(16) _Float16 QKh[64 * 64];
  __shared__ float    bvec[64];

  const int bh = blockIdx.x;                     // 0..95
  const int b = bh / NH, h = bh % NH;
  const int tid = threadIdx.x, wave = tid >> 5, lane = tid & 31;
  const int n_l = lane & 15, mbase = wave * 16 + (lane >> 4) * 8;

  for (int i = tid; i < 64 * 64; i += 128) { Sf[i] = 0.f; Sh[i] = (_Float16)0.f; }
  __syncthreads();

  const size_t baseQK = (size_t)bh * LQ * HD;
  v8f zero = {};

  for (int ci = 0; ci < NCHK; ++ci) {
    // ---- async stage chunk tiles (global -> LDS, 128b) ----------------
    {
      const uint4* qs = (const uint4*)(Qh + baseQK + (size_t)ci * 64 * 64);
      const uint4* ks = (const uint4*)(Kh + baseQK + (size_t)ci * 64 * 64);
      const uint4* vs = (const uint4*)(Vh + baseQK + (size_t)ci * 64 * 64);
      uint4* qd = (uint4*)Qis; uint4* kd = (uint4*)Kis; uint4* vd = (uint4*)Vis;
      for (int i = tid; i < 512; i += 128) {
        async_copy_b128(qd + i, qs + i);
        async_copy_b128(kd + i, ks + i);
        async_copy_b128(vd + i, vs + i);
      }
      if (tid < 64) bvec[tid] = Beta[(size_t)bh * LQ + ci * 64 + tid];
      async_wait0();
    }
    __syncthreads();

    // transposed K copy so every K^T operand is unit-stride in LDS
    for (int i = tid; i < 64 * 64; i += 128) {
      int t = i >> 6, d = i & 63;
      KisT[d * 64 + t] = Kis[i];
    }
    __syncthreads();

    // ---- T = I + strict_lower(beta * K K^T) --------------------------
    {
      v16h a0 = load_a_frag(Kis, wave * 16, 0, 64, 1);
      v16h a1 = load_a_frag(Kis, wave * 16, 32, 64, 1);
#pragma unroll
      for (int nt = 0; nt < 4; ++nt) {
        v8f acc = zero;
        acc = wmma16(a0, load_b_frag(KisT, 0,  nt * 16, 64, 1), acc);
        acc = wmma16(a1, load_b_frag(KisT, 32, nt * 16, 64, 1), acc);
#pragma unroll
        for (int r = 0; r < 8; ++r) {
          int m = mbase + r, n = nt * 16 + n_l;
          Tm[m * 64 + n] = (m > n) ? bvec[m] * acc[r] : (m == n ? 1.f : 0.f);
        }
      }
    }
    // ---- rhs = beta * (V - K S) --------------------------------------
    {
      v16h a0 = load_a_frag(Kis, wave * 16, 0, 64, 1);
      v16h a1 = load_a_frag(Kis, wave * 16, 32, 64, 1);
#pragma unroll
      for (int nt = 0; nt < 4; ++nt) {
        v8f acc = zero;
        acc = wmma16(a0, load_b_frag(Sh, 0,  nt * 16, 64, 1), acc);
        acc = wmma16(a1, load_b_frag(Sh, 32, nt * 16, 64, 1), acc);
#pragma unroll
        for (int r = 0; r < 8; ++r) {
          int m = mbase + r, n = nt * 16 + n_l;
          Rm[m * 64 + n] = bvec[m] * ((float)Vis[m * 64 + n] - acc[r]);
        }
      }
    }
    __syncthreads();

    // ---- forward substitution (unit lower-tri); thread = column ------
    if (tid < 64) {
      const int c = tid;
      for (int r = 1; r < 64; ++r) {
        float s = 0.f;
        for (int j = 0; j < r; ++j) s += Tm[r * 64 + j] * Rm[j * 64 + c];
        Rm[r * 64 + c] -= s;
      }
    }
    __syncthreads();
    for (int i = tid; i < 512; i += 128) {   // U -> f16, vectorized
      int m = i >> 3, jb = (i & 7) * 8;
      __align__(16) _Float16 tmp[8];
#pragma unroll
      for (int j = 0; j < 8; ++j) tmp[j] = (_Float16)Rm[m * 64 + jb + j];
      *(uint4*)&Uh[m * 64 + jb] = *(const uint4*)tmp;
    }
    __syncthreads();

    // ---- QK masked inclusive-lower -----------------------------------
    {
      v16h a0 = load_a_frag(Qis, wave * 16, 0, 64, 1);
      v16h a1 = load_a_frag(Qis, wave * 16, 32, 64, 1);
#pragma unroll
      for (int nt = 0; nt < 4; ++nt) {
        v8f acc = zero;
        acc = wmma16(a0, load_b_frag(KisT, 0,  nt * 16, 64, 1), acc);
        acc = wmma16(a1, load_b_frag(KisT, 32, nt * 16, 64, 1), acc);
#pragma unroll
        for (int r = 0; r < 8; ++r) {
          int m = mbase + r, n = nt * 16 + n_l;
          QKh[m * 64 + n] = (m >= n) ? (_Float16)acc[r] : (_Float16)0.f;
        }
      }
    }
    __syncthreads();

    // ---- O = Q S + tril(QK) U ; stage in Tm (free now), store 128b ----
    {
      v16h a0 = load_a_frag(Qis, wave * 16, 0, 64, 1);
      v16h a1 = load_a_frag(Qis, wave * 16, 32, 64, 1);
      v16h c0 = load_a_frag(QKh, wave * 16, 0, 64, 1);
      v16h c1 = load_a_frag(QKh, wave * 16, 32, 64, 1);
#pragma unroll
      for (int nt = 0; nt < 4; ++nt) {
        v8f acc = zero;
        acc = wmma16(a0, load_b_frag(Sh, 0,  nt * 16, 64, 1), acc);
        acc = wmma16(a1, load_b_frag(Sh, 32, nt * 16, 64, 1), acc);
        acc = wmma16(c0, load_b_frag(Uh, 0,  nt * 16, 64, 1), acc);
        acc = wmma16(c1, load_b_frag(Uh, 32, nt * 16, 64, 1), acc);
#pragma unroll
        for (int r = 0; r < 8; ++r)
          Tm[(mbase + r) * 64 + nt * 16 + n_l] = acc[r];
      }
    }
    __syncthreads();
    {
      _Float16* op = Oh + ((size_t)b * LQ + ci * 64) * CDIM + h * HD;
      for (int i = tid; i < 512; i += 128) {
        int m = i >> 3, jb = (i & 7) * 8;
        __align__(16) _Float16 tmp[8];
#pragma unroll
        for (int j = 0; j < 8; ++j) tmp[j] = (_Float16)Tm[m * 64 + jb + j];
        *(uint4*)&op[(size_t)m * CDIM + jb] = *(const uint4*)tmp;
      }
    }

    // ---- S += K^T U  (A = KisT row-major -> unit stride) --------------
    {
      v16h a0 = load_a_frag(KisT, wave * 16, 0, 64, 1);
      v16h a1 = load_a_frag(KisT, wave * 16, 32, 64, 1);
#pragma unroll
      for (int nt = 0; nt < 4; ++nt) {
        v8f acc = zero;
        acc = wmma16(a0, load_b_frag(Uh, 0,  nt * 16, 64, 1), acc);
        acc = wmma16(a1, load_b_frag(Uh, 32, nt * 16, 64, 1), acc);
#pragma unroll
        for (int r = 0; r < 8; ++r) {
          int m = mbase + r, n = nt * 16 + n_l;
          float nv = Sf[m * 64 + n] + acc[r];
          Sf[m * 64 + n] = nv;
          Sh[m * 64 + n] = (_Float16)nv;
        }
      }
    }
    __syncthreads();
  }
}

// ---------------------------------------------------------------------------
// y = o @ Wo ; out = t + y, stored transposed as [B,C,H,W] fp32.
// ---------------------------------------------------------------------------
__global__ __launch_bounds__(128) void out_kernel(
    const _Float16* __restrict__ Oh, const _Float16* __restrict__ Woh,
    const float* __restrict__ x, float* __restrict__ out) {
  __shared__ __align__(16) _Float16 As[64 * 32];   // [m][k]
  __shared__ __align__(16) _Float16 Bs[32 * 64];   // [k][n]

  const int b   = blockIdx.x >> 6;
  const int l0  = (blockIdx.x & 63) << 6;
  const int n0g = blockIdx.y * 64;
  const int tid = threadIdx.x, wave = tid >> 5, lane = tid & 31;
  const int n_l = lane & 15, mbase = wave * 16 + (lane >> 4) * 8;

  const _Float16* Ob = Oh + ((size_t)b * LQ + l0) * CDIM;
  v8f zero = {};
  v8f acc[4] = {zero, zero, zero, zero};

  for (int k0 = 0; k0 < CDIM; k0 += 32) {
    // A tile: 64 rows x 4 uint4 each
    {
      uint4* adst = (uint4*)As;
      for (int i = tid; i < 256; i += 128) {
        int m = i >> 2, j = i & 3;
        adst[i] = *(const uint4*)(Ob + (size_t)m * CDIM + k0 + j * 8);
      }
    }
    {
      const uint4* wsrc = (const uint4*)(Woh + (size_t)k0 * CDIM + n0g);
      uint4* bdst = (uint4*)Bs;
      for (int i = tid; i < 256; i += 128) {
        int kk = i >> 3, j = i & 7;
        bdst[i] = wsrc[(size_t)kk * (CDIM / 8) + j];
      }
    }
    __syncthreads();
    v16h a = load_a_frag(As, wave * 16, 0, /*rs=*/32, /*cs=*/1);
#pragma unroll
    for (int nt = 0; nt < 4; ++nt)
      acc[nt] = wmma16(a, load_b_frag(Bs, 0, nt * 16, 64, 1), acc[nt]);
    __syncthreads();
  }

  // residual add + transposed store: each lane owns 8 consecutive l's at
  // one channel n -> two float4 ops.
#pragma unroll
  for (int nt = 0; nt < 4; ++nt) {
    int n = n0g + nt * 16 + n_l;
    size_t idx = ((size_t)b * CDIM + n) * LQ + l0 + mbase;
    float4 x0 = *(const float4*)&x[idx];
    float4 x1 = *(const float4*)&x[idx + 4];
    float4 y0, y1;
    y0.x = x0.x + acc[nt][0]; y0.y = x0.y + acc[nt][1];
    y0.z = x0.z + acc[nt][2]; y0.w = x0.w + acc[nt][3];
    y1.x = x1.x + acc[nt][4]; y1.y = x1.y + acc[nt][5];
    y1.z = x1.z + acc[nt][6]; y1.w = x1.w + acc[nt][7];
    *(float4*)&out[idx]     = y0;
    *(float4*)&out[idx + 4] = y1;
  }
}

// ---------------------------------------------------------------------------
extern "C" void kernel_launch(void* const* d_in, const int* in_sizes, int n_in,
                              void* d_out, int out_size, void* d_ws, size_t ws_size,
                              hipStream_t stream) {
  const float* x  = (const float*)d_in[0];
  const float* Wq = (const float*)d_in[1];
  const float* Wk = (const float*)d_in[2];
  const float* Wv = (const float*)d_in[3];
  const float* Wb = (const float*)d_in[4];
  const float* Wo = (const float*)d_in[5];
  float* out = (float*)d_out;

  const size_t nQKV = (size_t)BQ * NH * LQ * HD;   // 25,165,824
  const size_t nO   = (size_t)BQ * LQ * CDIM;      // 25,165,824
  const size_t nB   = (size_t)BQ * NH * LQ;        // 393,216
  const size_t nW   = (size_t)CDIM * CDIM;         // 589,824

  _Float16* Qh   = (_Float16*)d_ws;
  _Float16* Kh   = Qh + nQKV;
  _Float16* Vh   = Kh + nQKV;
  _Float16* Ohws = Vh + nQKV;
  float*    Beta = (float*)(Ohws + nO);
  _Float16* Wqh  = (_Float16*)(Beta + nB);
  _Float16* Wkh  = Wqh + nW;
  _Float16* Wvh  = Wkh + nW;
  _Float16* Woh  = Wvh + nW;

  const int cvtBlocks = (int)((nW + 255) / 256);
  cvt_kernel<<<cvtBlocks, 256, 0, stream>>>(Wq, Wqh, (int)nW);
  cvt_kernel<<<cvtBlocks, 256, 0, stream>>>(Wk, Wkh, (int)nW);
  cvt_kernel<<<cvtBlocks, 256, 0, stream>>>(Wv, Wvh, (int)nW);
  cvt_kernel<<<cvtBlocks, 256, 0, stream>>>(Wo, Woh, (int)nW);

  proj_kernel<<<dim3(BQ * (LQ / 64), NH, 3), 128, 0, stream>>>(
      x, Wqh, Wkh, Wvh, Qh, Kh, Vh);
  beta_kernel<<<dim3(LQ / 256, NH, BQ), 256, 0, stream>>>(x, Wb, Beta);
  delta_kernel<<<BQ * NH, 128, 0, stream>>>(Qh, Kh, Vh, Beta, Ohws);
  out_kernel<<<dim3(BQ * (LQ / 64), NH), 128, 0, stream>>>(Ohws, Woh, x, out);
}
